// PSP_Pooling_70042326663190
// MI455X (gfx1250) — matrix operand order repaired and checked by reading, not compiled, not run on hardware
//
#include <hip/hip_runtime.h>
#include <hip/hip_bf16.h>
#include <float.h>

typedef __attribute__((ext_vector_type(16))) _Float16 v16h;
typedef __attribute__((ext_vector_type(8)))  _Float16 v8h;
typedef __attribute__((ext_vector_type(8)))  float    v8f;

#define Bn 4
#define Cn 128
#define Hn 256
#define Wn 256
#define HW (Hn * Wn)
#define FCW 640            // (DEPTH+1)*C
#define EPS_BN 1e-5f
#define NGEMM_WG 4096      // B * 64 blocks * 16 row-pairs
#define LDSW 136           // 128 channels + 8-half pad (keeps 16B align, rotates banks)

// ---------------------------------------------------------------------------
// Kernel 0: split W0 = final_w[:, :128] into hi/lo f16 once (removes all
// per-use conversion VALU from the GEMM inner loop).
// ---------------------------------------------------------------------------
__global__ __launch_bounds__(256)
void psp_wsplit(const float* __restrict__ fw, _Float16* __restrict__ whi,
                _Float16* __restrict__ wlo)
{
    const int idx = blockIdx.x * 256 + threadIdx.x;   // 128*128
    const int o = idx >> 7, k = idx & 127;
    float v = fw[(long)o * FCW + k];
    _Float16 hi = (_Float16)v;
    whi[idx] = hi;
    wlo[idx] = (_Float16)(v - (float)hi);
}

// ---------------------------------------------------------------------------
// Kernel 1: per-(b,c) hierarchical block maxima (8x8 / 4x4 / 2x2 / 1x1 grids)
// ---------------------------------------------------------------------------
__global__ __launch_bounds__(256)
void psp_blockmax(const float* __restrict__ x, float* __restrict__ pm8,
                  float* __restrict__ pm4, float* __restrict__ pm2,
                  float* __restrict__ pm1)
{
    __shared__ float rowmax[256 * 8];
    __shared__ float l8[64];
    __shared__ float l4[16];
    __shared__ float l2[4];
    const int bc = blockIdx.x;          // b*128 + c
    const int t  = threadIdx.x;         // one row per thread
    const float* row = x + (long)bc * HW + (long)t * Wn;
#pragma unroll
    for (int g = 0; g < 8; ++g) {
        float m = -FLT_MAX;
        for (int i = 0; i < 32; ++i) m = fmaxf(m, row[g * 32 + i]);
        rowmax[t * 8 + g] = m;
    }
    __syncthreads();
    if (t < 64) {                        // 8x8 grid (32x32 blocks)
        int gy = t >> 3, gx = t & 7;
        float mm = -FLT_MAX;
        for (int r = 0; r < 32; ++r) mm = fmaxf(mm, rowmax[(gy * 32 + r) * 8 + gx]);
        pm8[(long)bc * 64 + t] = mm;
        l8[t] = mm;
    }
    __syncthreads();
    if (t < 16) {                        // 4x4 grid
        int gy = t >> 2, gx = t & 3;
        float mm = fmaxf(fmaxf(l8[(gy*2)*8 + gx*2],     l8[(gy*2)*8 + gx*2 + 1]),
                         fmaxf(l8[(gy*2+1)*8 + gx*2],   l8[(gy*2+1)*8 + gx*2 + 1]));
        pm4[(long)bc * 16 + t] = mm;
        l4[t] = mm;
    }
    __syncthreads();
    if (t < 4) {                         // 2x2 grid
        int gy = t >> 1, gx = t & 1;
        float mm = fmaxf(fmaxf(l4[(gy*2)*4 + gx*2],     l4[(gy*2)*4 + gx*2 + 1]),
                         fmaxf(l4[(gy*2+1)*4 + gx*2],   l4[(gy*2+1)*4 + gx*2 + 1]));
        pm2[(long)bc * 4 + t] = mm;
        l2[t] = mm;
    }
    __syncthreads();
    if (t == 0) pm1[bc] = fmaxf(fmaxf(l2[0], l2[1]), fmaxf(l2[2], l2[3]));
}

// ---------------------------------------------------------------------------
// Kernel 2: pooled-branch 1x1 conv on the tiny grids:
//   yd[d][b][o][g] = sum_c conv_w[d][o][c] * pm_d[b][c][g]
// ---------------------------------------------------------------------------
__global__ __launch_bounds__(256)
void psp_pool_conv(const float* __restrict__ conv_w,
                   const float* __restrict__ pm8, const float* __restrict__ pm4,
                   const float* __restrict__ pm2, const float* __restrict__ pm1,
                   float* __restrict__ yd)
{
    const int idx = blockIdx.x * 256 + threadIdx.x;   // 4*4*128*64 = 131072
    const int g = idx & 63;
    const int o = (idx >> 6) & 127;
    const int b = (idx >> 13) & 3;
    const int d = idx >> 15;
    const int gcnt = 1 << (2 * d);
    if (g >= gcnt) return;
    const float* pm = (d == 0) ? pm1 : (d == 1) ? pm2 : (d == 2) ? pm4 : pm8;
    const float* wrow = conv_w + (long)(d * Cn + o) * Cn;
    float acc = 0.f;
    for (int c = 0; c < Cn; ++c) acc += wrow[c] * pm[(long)(b * Cn + c) * gcnt + g];
    yd[((long)(d * Bn + b) * Cn + o) * 64 + g] = acc;
}

// ---------------------------------------------------------------------------
// Kernel 3: BN on pooled branches. Block-broadcast means stats over (B,H,W)
// equal stats over (b, grid-cells) (equal-area blocks).
// ---------------------------------------------------------------------------
__global__ __launch_bounds__(256)
void psp_pool_bn(const float* __restrict__ yd, const float* __restrict__ conv_g,
                 const float* __restrict__ conv_b, float* __restrict__ pd)
{
    const int idx = blockIdx.x * 256 + threadIdx.x;   // 512 = DEPTH*C
    if (idx >= 512) return;
    const int d = idx >> 7, o = idx & 127;
    const int gcnt = 1 << (2 * d);
    const int n = Bn * gcnt;
    float sum = 0.f, ss = 0.f;
    for (int b = 0; b < Bn; ++b)
        for (int g = 0; g < gcnt; ++g) {
            float v = yd[((long)(d * Bn + b) * Cn + o) * 64 + g];
            sum += v; ss += v * v;
        }
    float mean = sum / (float)n;
    float var  = fmaxf(ss / (float)n - mean * mean, 0.f);
    float s = conv_g[d * Cn + o] * rsqrtf(var + EPS_BN);
    float t0 = conv_b[d * Cn + o] - mean * s;
    for (int b = 0; b < Bn; ++b)
        for (int g = 0; g < gcnt; ++g) {
            long i = ((long)(d * Bn + b) * Cn + o) * 64 + g;
            pd[i] = yd[i] * s + t0;
        }
}

// ---------------------------------------------------------------------------
// Kernel 4: fold pooled branches through final_w[:,128:] into a per-32x32-block
// additive term: add[b][o][blk] (blk over the 8x8 base grid).
// ---------------------------------------------------------------------------
__global__ __launch_bounds__(256)
void psp_addbias(const float* __restrict__ fw, const float* __restrict__ pd,
                 float* __restrict__ addb)
{
    const int idx = blockIdx.x * 256 + threadIdx.x;   // 4*128*64 = 32768
    const int blk = idx & 63;
    const int o   = (idx >> 6) & 127;
    const int b   = idx >> 13;
    const int gy = blk >> 3, gx = blk & 7;
    float acc = 0.f;
    for (int d = 0; d < 4; ++d) {
        const int sh = 3 - d;
        const int gd = (gy >> sh) * (1 << d) + (gx >> sh);
        const float* wrow = fw + (long)o * FCW + (1 + d) * Cn;
        const float* prow = pd + (long)(d * Bn + b) * Cn * 64;
        for (int c = 0; c < Cn; ++c) acc += wrow[c] * prow[(long)c * 64 + gd];
    }
    addb[((long)(b * Cn + o) << 6) + blk] = acc;
}

// ---------------------------------------------------------------------------
// Kernel 5 (used twice): main 128x128 1x1 conv via split-f16 WMMA.
//   pass=0: compute y = W0*x + add, reduce per-channel sum/sumsq -> partials
//   pass=1: recompute y, apply precomputed BN scale/shift, write out
// Tile: M=128 (8 waves x 16 rows), N=64 spatial (2 rows of 32), K=128.
// ---------------------------------------------------------------------------
__global__ __launch_bounds__(256)
void psp_main_gemm(const float* __restrict__ x,
                   const _Float16* __restrict__ whi, const _Float16* __restrict__ wlo,
                   const float* __restrict__ addb, const float* __restrict__ scale,
                   const float* __restrict__ shiftb, float* __restrict__ partials,
                   float* __restrict__ out, int pass)
{
    __shared__ __align__(32) _Float16 sxh[64 * LDSW];
    __shared__ __align__(32) _Float16 sxl[64 * LDSW];
    __shared__ float redS[128 * 16];
    __shared__ float redQ[128 * 16];

    const int tid  = threadIdx.x;
    const int lane = tid & 31;
    const int wv   = tid >> 5;

    const int wg  = blockIdx.x;
    const int sub = wg & 15;
    const int blk = (wg >> 4) & 63;
    const int b   = wg >> 10;
    const int hb  = ((blk >> 3) << 5) + (sub << 1);
    const int wb  = (blk & 7) << 5;

    // Stage x tile (128 ch x 64 pos) as split hi/lo f16, pos-major + pad.
    // Quad-channel packing -> one ds_store_b64 per 4 halves.
    {
        const float* xb = x + (long)(b * Cn) * HW + (long)hb * Wn + wb;
        for (int it = tid; it < 64 * 32; it += 256) {   // (pos, channel-quad)
            const int p  = it & 63;
            const int cq = it >> 6;                     // 0..31 -> c0 = 4*cq
            const float* px = xb + (long)(cq * 4) * HW + (p >> 5) * Wn + (p & 31);
            union { _Float16 h[4]; unsigned long long u; } uh, ul;
#pragma unroll
            for (int e = 0; e < 4; ++e) {
                float v = px[(long)e * HW];
                _Float16 hi = (_Float16)v;
                uh.h[e] = hi;
                ul.h[e] = (_Float16)(v - (float)hi);
            }
            *(unsigned long long*)&sxh[p * LDSW + cq * 4] = uh.u;
            *(unsigned long long*)&sxl[p * LDSW + cq * 4] = ul.u;
        }
    }

    // A fragments: wave's 16 rows of pre-split W0, 4 K-chunks, hi/lo.
    // ISA 16-bit A 16x32 layout: lanes0-15 K=0..7,16..23 ; lanes16-31 K=8..15,24..31.
    const int m0    = wv << 4;
    const int am    = lane & 15;
    const int kbase = (lane < 16) ? 0 : 8;
    const _Float16* wh = whi + (long)(m0 + am) * Cn + kbase;
    const _Float16* wl = wlo + (long)(m0 + am) * Cn + kbase;
    v16h a_hi[4], a_lo[4];
#pragma unroll
    for (int q = 0; q < 4; ++q) {
        v8h ah0 = *(const v8h*)(wh + q * 32);
        v8h ah1 = *(const v8h*)(wh + q * 32 + 16);
        v8h al0 = *(const v8h*)(wl + q * 32);
        v8h al1 = *(const v8h*)(wl + q * 32 + 16);
#pragma unroll
        for (int i = 0; i < 8; ++i) {
            a_hi[q][i] = ah0[i]; a_hi[q][i + 8] = ah1[i];
            a_lo[q][i] = al0[i]; a_lo[q][i + 8] = al1[i];
        }
    }

    // Per-output-row constants (C/D layout: VGPR r -> M = r or r+8).
    const int obase = m0 + ((lane < 16) ? 0 : 8);
    float addv[8], sc[8], sf[8];
#pragma unroll
    for (int r = 0; r < 8; ++r) {
        const int o = obase + r;
        addv[r] = addb[((long)(b * Cn + o) << 6) + blk];
        sc[r] = pass ? scale[o]  : 0.f;
        sf[r] = pass ? shiftb[o] : 0.f;
    }

    __syncthreads();

    const int nlane = lane & 15;
    const int koff  = (lane < 16) ? 0 : 8;    // B 32x16 layout: lanes0-15 K=0..7,16..23 ; lanes16-31 K=8..15,24..31
    float ssum[8] = {0,0,0,0,0,0,0,0};
    float ssq [8] = {0,0,0,0,0,0,0,0};

#pragma unroll
    for (int j = 0; j < 4; ++j) {            // 4 N-groups of 16 positions
        v8f acc = {0.f,0.f,0.f,0.f,0.f,0.f,0.f,0.f};
        const int pos = (j << 4) + nlane;
#pragma unroll
        for (int q = 0; q < 4; ++q) {        // K chunks of 32 channels
            const _Float16* ph = &sxh[pos * LDSW + (q << 5) + koff];
            const _Float16* pl = &sxl[pos * LDSW + (q << 5) + koff];
            v8h bh0 = *(const v8h*)(ph);
            v8h bh1 = *(const v8h*)(ph + 16);
            v8h bl0 = *(const v8h*)(pl);
            v8h bl1 = *(const v8h*)(pl + 16);
            v16h bh, bl;
#pragma unroll
            for (int i = 0; i < 8; ++i) {
                bh[i] = bh0[i]; bh[i + 8] = bh1[i];
                bl[i] = bl0[i]; bl[i + 8] = bl1[i];
            }
            // split-precision: w_hi*x_hi + w_hi*x_lo + w_lo*x_hi, f32 accumulate
            acc = __builtin_amdgcn_wmma_f32_16x16x32_f16(false, a_hi[q], false, bh,
                                                         (short)0, acc, false, false);
            acc = __builtin_amdgcn_wmma_f32_16x16x32_f16(false, a_hi[q], false, bl,
                                                         (short)0, acc, false, false);
            acc = __builtin_amdgcn_wmma_f32_16x16x32_f16(false, a_lo[q], false, bh,
                                                         (short)0, acc, false, false);
        }
        if (pass) {
            const int hh = hb + (pos >> 5);
            const int ww = wb + (pos & 31);
            float* op = out + (long)(b * Cn + obase) * HW + (long)hh * Wn + ww;
#pragma unroll
            for (int r = 0; r < 8; ++r) {
                float y = acc[r] + addv[r];
                op[(long)r * HW] = y * sc[r] + sf[r];
            }
        } else {
#pragma unroll
            for (int r = 0; r < 8; ++r) {
                float y = acc[r] + addv[r];
                ssum[r] += y;
                ssq[r]  += y * y;
            }
        }
    }

    if (!pass) {
        // Deterministic tree reduction: slot (o, nlane) is written by exactly one lane.
#pragma unroll
        for (int r = 0; r < 8; ++r) {
            redS[(obase + r) * 16 + nlane] = ssum[r];
            redQ[(obase + r) * 16 + nlane] = ssq[r];
        }
        __syncthreads();
        if (tid < 128) {
            float s = 0.f, q = 0.f;
#pragma unroll
            for (int i = 0; i < 16; ++i) { s += redS[tid * 16 + i]; q += redQ[tid * 16 + i]; }
            partials[(long)wg * 128 + tid] = s;
            partials[(long)NGEMM_WG * 128 + (long)wg * 128 + tid] = q;
        }
    }
}

// ---------------------------------------------------------------------------
// Kernel 6: finalize BN stats of the big conv output.
// ---------------------------------------------------------------------------
__global__ void psp_finstats(const float* __restrict__ partials,
                             const float* __restrict__ final_g,
                             const float* __restrict__ final_b,
                             float* __restrict__ scale, float* __restrict__ shiftb)
{
    const int o = threadIdx.x;
    if (o >= 128) return;
    float sum = 0.f, ss = 0.f;
    for (int wg = 0; wg < NGEMM_WG; ++wg) {
        sum += partials[(long)wg * 128 + o];
        ss  += partials[(long)NGEMM_WG * 128 + (long)wg * 128 + o];
    }
    const float invn = 1.0f / (float)(Bn * HW);
    float mean = sum * invn;
    float var  = fmaxf(ss * invn - mean * mean, 0.f);
    float s = final_g[o] * rsqrtf(var + EPS_BN);
    scale[o]  = s;
    shiftb[o] = final_b[o] - mean * s;
}

// ---------------------------------------------------------------------------
extern "C" void kernel_launch(void* const* d_in, const int* in_sizes, int n_in,
                              void* d_out, int out_size, void* d_ws, size_t ws_size,
                              hipStream_t stream)
{
    (void)in_sizes; (void)n_in; (void)out_size; (void)ws_size;
    const float* x       = (const float*)d_in[0];
    const float* conv_w  = (const float*)d_in[1];
    const float* conv_g  = (const float*)d_in[2];
    const float* conv_b  = (const float*)d_in[3];
    const float* final_w = (const float*)d_in[4];
    const float* final_g = (const float*)d_in[5];
    const float* final_b = (const float*)d_in[6];
    float* out = (float*)d_out;
    float* ws  = (float*)d_ws;

    // Workspace layout (floats): ~5.6 MB total.
    float* pm8      = ws;                  // 4*128*64
    float* pm4      = pm8  + 32768;        // 4*128*16
    float* pm2      = pm4  + 8192;         // 4*128*4
    float* pm1      = pm2  + 2048;         // 4*128
    float* yd       = pm1  + 512;          // 4*4*128*64
    float* pd       = yd   + 131072;       // 4*4*128*64
    float* addb     = pd   + 131072;       // 4*128*64
    float* scale    = addb + 32768;        // 128
    float* shiftb   = scale + 128;         // 128
    float* partials = shiftb + 128;        // 2*4096*128
    _Float16* whi   = (_Float16*)(partials + 2L * NGEMM_WG * 128);  // 128*128 halves
    _Float16* wlo   = whi + Cn * Cn;                                 // 128*128 halves

    psp_wsplit   <<<64,  256, 0, stream>>>(final_w, whi, wlo);
    psp_blockmax <<<Bn * Cn, 256, 0, stream>>>(x, pm8, pm4, pm2, pm1);
    psp_pool_conv<<<512, 256, 0, stream>>>(conv_w, pm8, pm4, pm2, pm1, yd);
    psp_pool_bn  <<<2,   256, 0, stream>>>(yd, conv_g, conv_b, pd);
    psp_addbias  <<<128, 256, 0, stream>>>(final_w, pd, addb);
    psp_main_gemm<<<NGEMM_WG, 256, 0, stream>>>(x, whi, wlo, addb, scale, shiftb,
                                                partials, out, 0);
    psp_finstats <<<1, 128, 0, stream>>>(partials, final_g, final_b, scale, shiftb);
    psp_main_gemm<<<NGEMM_WG, 256, 0, stream>>>(x, whi, wlo, addb, scale, shiftb,
                                                partials, out, 1);
}
